// AttentionBlock_64845416235442
// MI455X (gfx1250) — matrix operand order repaired and checked by reading, not compiled, not run on hardware
//
#include <hip/hip_runtime.h>
#include <hip/hip_bf16.h>

#define C_DIM 512
#define S_DIM 1024
#define BATCH 8
#define NHEAD 8
#define DHEAD 64
#define NGROUP 32
#define GSIZE 16
#define EPSV 1e-5f

typedef __bf16 bf16x16 __attribute__((ext_vector_type(16)));
typedef float  f32x8   __attribute__((ext_vector_type(8)));

union FragAB { uint4 u[2]; bf16x16 v; };

__device__ __forceinline__ unsigned short f2bf(float f) {
  unsigned u = __float_as_uint(f);
  unsigned r = (u + 0x7FFFu + ((u >> 16) & 1u)) >> 16;
  return (unsigned short)r;
}

// ---------------------------------------------------------------------------
// GroupNorm: x[B,C,H,W] f32 -> tokens t[B*S, C] bf16 (token-major, C contig)
// One block per (batch, group): 16 channels x 1024 pixels.
// ---------------------------------------------------------------------------
__global__ __launch_bounds__(256) void gn_kernel(const float* __restrict__ x,
                                                 const float* __restrict__ w,
                                                 const float* __restrict__ bgn,
                                                 unsigned short* __restrict__ t) {
  const int blk = blockIdx.x;
  const int b = blk / NGROUP, g = blk % NGROUP;
  const int tid = threadIdx.x;
  const size_t base = ((size_t)b * C_DIM + g * GSIZE) * S_DIM;

  float s0 = 0.f, s1 = 0.f;
  for (int i = tid; i < GSIZE * S_DIM; i += 256) {
    float v = x[base + i];
    s0 += v; s1 += v * v;
  }
  __shared__ float r0[256], r1[256];
  r0[tid] = s0; r1[tid] = s1;
  __syncthreads();
  for (int off = 128; off > 0; off >>= 1) {
    if (tid < off) { r0[tid] += r0[tid + off]; r1[tid] += r1[tid + off]; }
    __syncthreads();
  }
  const float invN = 1.0f / (float)(GSIZE * S_DIM);
  const float mean = r0[0] * invN;
  const float var  = r1[0] * invN - mean * mean;
  const float inv  = rsqrtf(var + EPSV);

  for (int i = tid; i < GSIZE * S_DIM; i += 256) {
    int cl = i >> 10;
    int p  = i & (S_DIM - 1);
    int c  = g * GSIZE + cl;
    float v = (x[base + i] - mean) * inv * w[c] + bgn[c];
    t[((size_t)b * S_DIM + p) * C_DIM + c] = f2bf(v);
  }
}

// ---------------------------------------------------------------------------
// Weight conversion f32 -> bf16
// ---------------------------------------------------------------------------
__global__ void cvt_kernel(const float* __restrict__ src,
                           unsigned short* __restrict__ dst, int n) {
  int i = blockIdx.x * blockDim.x + threadIdx.x;
  if (i < n) dst[i] = f2bf(src[i]);
}

// ---------------------------------------------------------------------------
// GEMM: Y[M,512] = A[M,512] @ W[512,512]^T + bias.  A,W bf16, accum f32.
// Wave tile 64x64 = 4x4 WMMA accumulators, K loop 512/32.
// mode 0: store bf16 [M,C]                       (Q, K)
// mode 1: store bf16 transposed [(b*C+n)*S + s]  (V^T for attention)
// mode 2: store f32  NCHW + bias + residual      (output projection)
// ---------------------------------------------------------------------------
__global__ __launch_bounds__(256) void gemm_kernel(
    const unsigned short* __restrict__ A, const unsigned short* __restrict__ W,
    const float* __restrict__ bias, unsigned short* __restrict__ outB,
    float* __restrict__ outF, const float* __restrict__ resid, int mode) {
  const int waveId = blockIdx.x * 8 + (threadIdx.x >> 5);
  const int tM = waveId >> 3;        // 128 M tiles
  const int tN = waveId & 7;         // 8 N tiles
  const int lane = threadIdx.x & 31;
  const int r = lane & 15, h = lane >> 4;
  const int m0 = tM << 6, n0 = tN << 6;

  f32x8 acc[4][4];
#pragma unroll
  for (int mi = 0; mi < 4; mi++)
#pragma unroll
    for (int ni = 0; ni < 4; ni++)
#pragma unroll
      for (int j = 0; j < 8; j++) acc[mi][ni][j] = 0.0f;

  for (int k0 = 0; k0 < C_DIM; k0 += 32) {
    FragAB a[4], bb[4];
#pragma unroll
    for (int mi = 0; mi < 4; mi++) {
      // A frag (16x32): lane r holds row m0+mi*16+r, K = h*8..h*8+7 and +16
      const unsigned short* p = A + (size_t)(m0 + mi * 16 + r) * C_DIM + k0 + h * 8;
      a[mi].u[0] = *(const uint4*)p;
      a[mi].u[1] = *(const uint4*)(p + 16);
    }
#pragma unroll
    for (int ni = 0; ni < 4; ni++) {
      // B frag (32x16): lane r = column n, holds K = h*8..h*8+7 and +16
      const unsigned short* p = W + (size_t)(n0 + ni * 16 + r) * C_DIM + k0 + h * 8;
      bb[ni].u[0] = *(const uint4*)p;
      bb[ni].u[1] = *(const uint4*)(p + 16);
    }
#pragma unroll
    for (int mi = 0; mi < 4; mi++)
#pragma unroll
      for (int ni = 0; ni < 4; ni++)
        acc[mi][ni] = __builtin_amdgcn_wmma_f32_16x16x32_bf16(
            false, a[mi].v, false, bb[ni].v, (short)0, acc[mi][ni], false, false);
  }

#pragma unroll
  for (int mi = 0; mi < 4; mi++)
#pragma unroll
    for (int ni = 0; ni < 4; ni++)
#pragma unroll
      for (int j = 0; j < 8; j++) {
        int m = m0 + mi * 16 + h * 8 + j;
        int n = n0 + ni * 16 + r;
        float v = acc[mi][ni][j] + bias[n];
        if (mode == 0) {
          outB[(size_t)m * C_DIM + n] = f2bf(v);
        } else if (mode == 1) {
          int b = m >> 10, s = m & 1023;
          outB[((size_t)b * C_DIM + n) * S_DIM + s] = f2bf(v);
        } else {
          int b = m >> 10, s = m & 1023;
          size_t o = ((size_t)b * C_DIM + n) * S_DIM + s;
          outF[o] = v + resid[o];
        }
      }
}

// ---------------------------------------------------------------------------
// Flash attention: per (batch, head), each wave owns 16 queries.
// q,k: bf16 [B*S, C] token-major.  vt: bf16 [(b*C+c), S] key-contiguous.
// Online softmax over key tiles of 32; P transposed D->A layout via LDS.
// ---------------------------------------------------------------------------
__global__ __launch_bounds__(256) void attn_kernel(
    const unsigned short* __restrict__ q, const unsigned short* __restrict__ k,
    const unsigned short* __restrict__ vt, unsigned short* __restrict__ o) {
  __shared__ __align__(16) unsigned short plds[8][16][32];

  const int wave = threadIdx.x >> 5;
  const int lane = threadIdx.x & 31;
  const int r = lane & 15, h = lane >> 4;
  const int bh = blockIdx.x >> 3, qblk = blockIdx.x & 7;
  const int b = bh >> 3, hd = bh & 7;
  const int q0 = qblk * 128 + wave * 16;
  const float scl = 0.125f;  // 1/sqrt(64)

  // Preload Q A-fragments for both dH halves (persist over key loop)
  FragAB qa[2];
#pragma unroll
  for (int dk = 0; dk < 2; dk++) {
    const unsigned short* p =
        q + ((size_t)b * S_DIM + q0 + r) * C_DIM + hd * DHEAD + dk * 32 + h * 8;
    qa[dk].u[0] = *(const uint4*)p;
    qa[dk].u[1] = *(const uint4*)(p + 16);
  }

  f32x8 o_acc[4];
#pragma unroll
  for (int dt = 0; dt < 4; dt++)
#pragma unroll
    for (int j = 0; j < 8; j++) o_acc[dt][j] = 0.0f;
  float mrow[8], lrow[8];
#pragma unroll
  for (int j = 0; j < 8; j++) { mrow[j] = -1e30f; lrow[j] = 0.0f; }

  for (int kt = 0; kt < S_DIM; kt += 32) {
    // ---- S = Q @ K^T for two 16-key subtiles ----
    f32x8 s[2];
#pragma unroll
    for (int st = 0; st < 2; st++) {
#pragma unroll
      for (int j = 0; j < 8; j++) s[st][j] = 0.0f;
      const unsigned short* p =
          k + ((size_t)b * S_DIM + kt + st * 16 + r) * C_DIM + hd * DHEAD + h * 8;
      FragAB kb;
      kb.u[0] = *(const uint4*)p;
      kb.u[1] = *(const uint4*)(p + 16);
      s[st] = __builtin_amdgcn_wmma_f32_16x16x32_bf16(false, qa[0].v, false, kb.v,
                                                      (short)0, s[st], false, false);
      kb.u[0] = *(const uint4*)(p + 32);
      kb.u[1] = *(const uint4*)(p + 48);
      s[st] = __builtin_amdgcn_wmma_f32_16x16x32_bf16(false, qa[1].v, false, kb.v,
                                                      (short)0, s[st], false, false);
    }

    // ---- Online softmax (row stats live in register index j, half h) ----
#pragma unroll
    for (int j = 0; j < 8; j++) {
      float a0 = s[0][j] * scl, a1 = s[1][j] * scl;
      float mx = fmaxf(a0, a1);
      mx = fmaxf(mx, __shfl_xor(mx, 1, 32));
      mx = fmaxf(mx, __shfl_xor(mx, 2, 32));
      mx = fmaxf(mx, __shfl_xor(mx, 4, 32));
      mx = fmaxf(mx, __shfl_xor(mx, 8, 32));
      float mnew = fmaxf(mrow[j], mx);
      float corr = __expf(mrow[j] - mnew);
      float p0 = __expf(a0 - mnew);
      float p1 = __expf(a1 - mnew);
      float rs = p0 + p1;
      rs += __shfl_xor(rs, 1, 32);
      rs += __shfl_xor(rs, 2, 32);
      rs += __shfl_xor(rs, 4, 32);
      rs += __shfl_xor(rs, 8, 32);
      lrow[j] = lrow[j] * corr + rs;
      mrow[j] = mnew;
#pragma unroll
      for (int dt = 0; dt < 4; dt++) o_acc[dt][j] *= corr;
      // D layout -> LDS row-major P[16q][32k]
      plds[wave][h * 8 + j][r]      = f2bf(p0);
      plds[wave][h * 8 + j][16 + r] = f2bf(p1);
    }
    // wave-local fence: LDS ops of one wave are in-order; force compiler order
    asm volatile("s_wait_dscnt 0" ::: "memory");

    // Reload P in A-fragment layout (lane r = query row, K = h*8..h*8+7 and +16)
    FragAB pa;
    {
      const unsigned short* p = &plds[wave][r][h * 8];
      pa.u[0] = *(const uint4*)p;
      pa.u[1] = *(const uint4*)(p + 16);
    }

    // ---- O += P @ V  (V^T layout gives key-contiguous B frags) ----
#pragma unroll
    for (int dt = 0; dt < 4; dt++) {
      const unsigned short* p =
          vt + ((size_t)b * C_DIM + hd * DHEAD + dt * 16 + r) * S_DIM + kt + h * 8;
      FragAB vb;
      vb.u[0] = *(const uint4*)p;
      vb.u[1] = *(const uint4*)(p + 16);
      o_acc[dt] = __builtin_amdgcn_wmma_f32_16x16x32_bf16(false, pa.v, false, vb.v,
                                                          (short)0, o_acc[dt], false, false);
    }
    asm volatile("s_wait_dscnt 0" ::: "memory");
  }

  // ---- Normalize and store attention output bf16 [M, C] ----
#pragma unroll
  for (int dt = 0; dt < 4; dt++)
#pragma unroll
    for (int j = 0; j < 8; j++) {
      float v = o_acc[dt][j] / lrow[j];
      size_t m = (size_t)b * S_DIM + q0 + h * 8 + j;
      int n = hd * DHEAD + dt * 16 + r;
      o[m * C_DIM + n] = f2bf(v);
    }
}

// ---------------------------------------------------------------------------
extern "C" void kernel_launch(void* const* d_in, const int* in_sizes, int n_in,
                              void* d_out, int out_size, void* d_ws, size_t ws_size,
                              hipStream_t stream) {
  const float* x    = (const float*)d_in[0];
  const float* gn_w = (const float*)d_in[1];
  const float* gn_b = (const float*)d_in[2];
  const float* wq   = (const float*)d_in[3];
  const float* bq   = (const float*)d_in[4];
  const float* wk   = (const float*)d_in[5];
  const float* bk   = (const float*)d_in[6];
  const float* wv   = (const float*)d_in[7];
  const float* bv   = (const float*)d_in[8];
  const float* wo   = (const float*)d_in[9];
  const float* bo   = (const float*)d_in[10];
  float* out = (float*)d_out;

  char* ws = (char*)d_ws;
  const size_t ACT = (size_t)8192 * 512 * 2;  // 8 MiB per bf16 activation buffer
  unsigned short* t   = (unsigned short*)(ws);
  unsigned short* qb  = (unsigned short*)(ws + 1 * ACT);
  unsigned short* kb  = (unsigned short*)(ws + 2 * ACT);
  unsigned short* vtb = (unsigned short*)(ws + 3 * ACT);
  unsigned short* ob  = (unsigned short*)(ws + 4 * ACT);
  unsigned short* wqb = (unsigned short*)(ws + 5 * ACT);
  unsigned short* wkb = wqb + 512 * 512;
  unsigned short* wvb = wkb + 512 * 512;
  unsigned short* wob = wvb + 512 * 512;

  gn_kernel<<<BATCH * NGROUP, 256, 0, stream>>>(x, gn_w, gn_b, t);
  cvt_kernel<<<1024, 256, 0, stream>>>(wq, wqb, 512 * 512);
  cvt_kernel<<<1024, 256, 0, stream>>>(wk, wkb, 512 * 512);
  cvt_kernel<<<1024, 256, 0, stream>>>(wv, wvb, 512 * 512);
  cvt_kernel<<<1024, 256, 0, stream>>>(wo, wob, 512 * 512);

  gemm_kernel<<<128, 256, 0, stream>>>(t, wqb, bq, qb, nullptr, nullptr, 0);
  gemm_kernel<<<128, 256, 0, stream>>>(t, wkb, bk, kb, nullptr, nullptr, 0);
  gemm_kernel<<<128, 256, 0, stream>>>(t, wvb, bv, vtb, nullptr, nullptr, 1);

  attn_kernel<<<BATCH * NHEAD * 8, 256, 0, stream>>>(qb, kb, vtb, ob);

  gemm_kernel<<<128, 256, 0, stream>>>(ob, wob, bo, nullptr, out, x, 2);
}
